// Model_10041633538519
// MI455X (gfx1250) — matrix operand order repaired and checked by reading, compiled, not run on hardware
//
#include <hip/hip_runtime.h>
#include <hip/hip_bf16.h>

#define T_LEN 36
#define FEATN 36
#define HID   128
#define BATCH 8192
#define KPAD  256      // padded K for gate GEMM: [0,108) x|m|d, [112,240) h, rest zero
#define HOFF  112
#define N4H   512
#define BT    32       // batch rows per block
#define NTHR  512      // 16 waves
#define BTF_  (BATCH * T_LEN * FEATN)

typedef __attribute__((ext_vector_type(16))) __bf16 v16bf;
typedef __attribute__((ext_vector_type(8)))  __bf16 bf16x8;
typedef __attribute__((ext_vector_type(8)))  float  v8f;

__device__ __forceinline__ float sigmoidf_(float x) { return 1.0f / (1.0f + expf(-x)); }

// ---------------------------------------------------------------------------
// Weight repack: Wcat bf16 [512][256], bias fold, W_hist split/pad, accum zero
// ---------------------------------------------------------------------------
__global__ void prep_kernel(const float* __restrict__ W_ih, const float* __restrict__ W_hh,
                            const float* __restrict__ b_ih, const float* __restrict__ b_hh,
                            const float* __restrict__ W_hist,
                            __bf16* __restrict__ Wcat, float* __restrict__ biasc,
                            __bf16* __restrict__ Whf, __bf16* __restrict__ Whb,
                            float* __restrict__ accum)
{
    int i = blockIdx.x * blockDim.x + threadIdx.x;
    int stride = gridDim.x * blockDim.x;
    for (int idx = i; idx < N4H * KPAD; idx += stride) {
        int n = idx / KPAD, k = idx % KPAD;
        float v = 0.0f;
        if (k < 108)                         v = W_ih[n * 108 + k];
        else if (k >= HOFF && k < HOFF+HID)  v = W_hh[n * HID + (k - HOFF)];
        Wcat[idx] = (__bf16)v;
    }
    for (int idx = i; idx < N4H; idx += stride) biasc[idx] = b_ih[idx] + b_hh[idx];
    for (int idx = i; idx < 48 * HID; idx += stride) {
        int f = idx / HID, k = idx % HID;
        float vf = 0.0f, vb = 0.0f;
        if (f < FEATN) { vf = W_hist[f * (2*HID) + k]; vb = W_hist[f * (2*HID) + HID + k]; }
        Whf[idx] = (__bf16)vf;
        Whb[idx] = (__bf16)vb;
    }
    for (int idx = i; idx < 2 * T_LEN; idx += stride) accum[idx] = 0.0f;
}

// ---------------------------------------------------------------------------
// LSTM over 36 steps; batch tile of 32 rows resident per block (16 waves).
// Wave w: M-stripe mt=w&1 (16 rows), hidden group q=w>>1 (16 cols). One tile
// per gate: nt = 8*g + q, so i/f/g/o of a (row,j) sit in the same lane and the
// cell update is register-only. Weight fragment loads are kept inside the
// t-loop via an opaque zero so LICM cannot hoist+spill them; they are served
// from L2 (one shared 256 KB copy for all blocks).
// ---------------------------------------------------------------------------
__global__ __launch_bounds__(NTHR) void lstm_kernel(
    const float* __restrict__ values, const float* __restrict__ masks,
    const float* __restrict__ deltas,
    const __bf16* __restrict__ Wcat, const float* __restrict__ biasc,
    const __bf16* __restrict__ Whist, float* __restrict__ xh_buf, int backward)
{
    __shared__ alignas(16) __bf16 U[BT][KPAD];   // 16 KB activation tile [x|m|d|h]
    __shared__ float bias_s[N4H];

    const int tid  = threadIdx.x;
    const int lane = tid & 31;
    const int wave = tid >> 5;                   // 0..15
    const int lo   = lane & 15;
    const int hi   = lane >> 4;
    const int mt   = wave & 1;                   // M tile (16 rows)
    const int q    = wave >> 1;                  // hidden 16-col group, 0..7
    const int b0   = blockIdx.x * BT;

    for (int idx = tid; idx < BT * KPAD; idx += NTHR) (&U[0][0])[idx] = (__bf16)0.0f;
    for (int idx = tid; idx < N4H; idx += NTHR) bias_s[idx] = biasc[idx];

    float c_reg[8];
    #pragma unroll
    for (int r = 0; r < 8; ++r) c_reg[r] = 0.0f;

    const int arow = 16 * mt + lo;
    const int j    = 16 * q + lo;                // hidden column this lane updates

    // hist output running pointer (waves 0..5 use it); advance +/-FEATN per t
    const int hmt = wave & 1;
    const int hnt = wave >> 1;
    const int hn  = 16 * hnt + lo;
    const int t0  = backward ? (T_LEN - 1) : 0;
    const int tstep = backward ? -FEATN : FEATN;
    float* xh_p = xh_buf +
        ((size_t)(b0 + 16 * hmt + 8 * hi) * T_LEN + t0) * FEATN + hn;

    for (int t = 0; t < T_LEN; ++t) {
        // opaque zero: defeats LICM of the (t-invariant) weight fragment loads
        int tzero;
        asm volatile("s_mov_b32 %0, 0" : "=s"(tzero));
        const __bf16* Wcat_t  = Wcat  + tzero;
        const __bf16* Whist_t = Whist + tzero;

        __syncthreads();                         // prev h writes visible
        // stage x|m|d for step t (fp32 -> bf16)
        for (int idx = tid; idx < BT * 108; idx += NTHR) {
            int row = idx / 108, col = idx % 108;
            size_t g = ((size_t)(b0 + row) * T_LEN + t) * FEATN;
            float v;
            if (col < 36)      v = values[g + col];
            else if (col < 72) v = masks[g + (col - 36)];
            else               v = deltas[g + (col - 72)];
            U[row][col] = (__bf16)v;
        }
        __syncthreads();

        // ---- hist_reg from current (pre-update) h: 6 waves, one 16x16 tile each
        if (wave < 6) {
            v8f hacc = {};
            const __bf16* hUp = &U[16 * hmt + lo][0];
            #pragma unroll
            for (int kk = 0; kk < 4; ++kk) {
                int ka = HOFF + 32 * kk + 8 * hi;
                bf16x8 a0 = *(const bf16x8*)(hUp + ka);
                bf16x8 a1 = *(const bf16x8*)(hUp + ka + 16);
                v16bf a;
                #pragma unroll
                for (int e = 0; e < 8; ++e) { a[e] = a0[e]; a[8 + e] = a1[e]; }
                const __bf16* Bp = Whist_t + (size_t)hn * HID + 32 * kk + 8 * hi;
                bf16x8 w0 = *(const bf16x8*)(Bp);
                bf16x8 w1 = *(const bf16x8*)(Bp + 16);
                v16bf b;
                #pragma unroll
                for (int e = 0; e < 8; ++e) { b[e] = w0[e]; b[8 + e] = w1[e]; }
                hacc = __builtin_amdgcn_wmma_f32_16x16x32_bf16(
                           false, a, false, b, (short)0, hacc, false, false);
            }
            if (hn < FEATN) {
                #pragma unroll
                for (int r = 0; r < 8; ++r)
                    xh_p[(size_t)r * T_LEN * FEATN] = hacc[r];
            }
        }
        xh_p += tstep;

        // ---- gate GEMM: [32 x 256] @ [256 x 512], K in 8 steps of 32
        v8f acc[4] = {};
        const __bf16* Up = &U[arow][0];
        #pragma unroll
        for (int kk = 0; kk < 8; ++kk) {
            int ka = 32 * kk + 8 * hi;
            bf16x8 a0 = *(const bf16x8*)(Up + ka);
            bf16x8 a1 = *(const bf16x8*)(Up + ka + 16);
            v16bf a;
            #pragma unroll
            for (int e = 0; e < 8; ++e) { a[e] = a0[e]; a[8 + e] = a1[e]; }
            #pragma unroll
            for (int g = 0; g < 4; ++g) {
                int nt = 8 * g + q;
                const __bf16* Bp = Wcat_t + (size_t)(16 * nt + lo) * KPAD + 32 * kk + 8 * hi;
                bf16x8 w0 = *(const bf16x8*)(Bp);
                bf16x8 w1 = *(const bf16x8*)(Bp + 16);
                v16bf b;
                #pragma unroll
                for (int e = 0; e < 8; ++e) { b[e] = w0[e]; b[8 + e] = w1[e]; }
                acc[g] = __builtin_amdgcn_wmma_f32_16x16x32_bf16(
                             false, a, false, b, (short)0, acc[g], false, false);
            }
        }
        __syncthreads();                         // all reads of U done before h overwrite

        // ---- in-register LSTM cell update; write new h back to U (bf16)
        {
            float bi = bias_s[j], bf = bias_s[128 + j];
            float bg = bias_s[256 + j], bo = bias_s[384 + j];
            #pragma unroll
            for (int r = 0; r < 8; ++r) {
                float gi = sigmoidf_(acc[0][r] + bi);
                float gf = sigmoidf_(acc[1][r] + bf);
                float gg = tanhf    (acc[2][r] + bg);
                float go = sigmoidf_(acc[3][r] + bo);
                float c  = gf * c_reg[r] + gi * gg;
                c_reg[r] = c;
                float h = go * tanhf(c);
                U[16 * mt + r + 8 * hi][HOFF + j] = (__bf16)h;
            }
        }
    }
}

// ---------------------------------------------------------------------------
// alpha = sigmoid([m|d] @ W_wc^T + b_wc); x_c = alpha*x_h + (1-alpha);
// imputations + evals passthrough + per-t masked-L1 partial sums
// ---------------------------------------------------------------------------
__global__ __launch_bounds__(256) void finalize_kernel(
    const float* __restrict__ fwd_values, const float* __restrict__ fwd_masks,
    const float* __restrict__ fwd_deltas, const float* __restrict__ fwd_evals,
    const float* __restrict__ fwd_eval_masks,
    const float* __restrict__ xhf, const float* __restrict__ xhb,
    const float* __restrict__ W_wc, const float* __restrict__ b_wc,
    const float* __restrict__ b_hist,
    float* __restrict__ out, float* __restrict__ accum)
{
    __shared__ float wc_s[FEATN * 2 * FEATN];
    __shared__ float bwc_s[FEATN];
    __shared__ float bh_s[FEATN];
    for (int idx = threadIdx.x; idx < FEATN * 2 * FEATN; idx += blockDim.x) wc_s[idx] = W_wc[idx];
    for (int idx = threadIdx.x; idx < FEATN; idx += blockDim.x) {
        bwc_s[idx] = b_wc[idx];
        bh_s[idx]  = b_hist[idx];
    }
    __syncthreads();

    int row = blockIdx.x * blockDim.x + threadIdx.x;
    if (row >= BATCH * T_LEN) return;
    int t = row % T_LEN;
    size_t base = (size_t)row * FEATN;

    float m[FEATN], dd[FEATN];
    #pragma unroll
    for (int f = 0; f < FEATN; ++f) { m[f] = fwd_masks[base + f]; dd[f] = fwd_deltas[base + f]; }

    float num = 0.0f, den = 0.0f;
    for (int f = 0; f < FEATN; ++f) {
        float z = bwc_s[f];
        const float* wr = &wc_s[f * 2 * FEATN];
        #pragma unroll
        for (int k = 0; k < FEATN; ++k) z += wr[k] * m[k];
        #pragma unroll
        for (int k = 0; k < FEATN; ++k) z += wr[FEATN + k] * dd[k];
        float alpha = sigmoidf_(z);
        float xh = xhf[base + f] + xhb[base + f] + bh_s[f];
        float xc = alpha * xh + (1.0f - alpha);
        out[1 + base + f] = xc;
        num += fabsf(fwd_values[base + f] - xc) * m[f];
        den += m[f];
        out[1 + (size_t)BTF_ + base + f]     = fwd_evals[base + f];
        out[1 + 2 * (size_t)BTF_ + base + f] = fwd_eval_masks[base + f];
    }
    atomicAdd(&accum[t], num);
    atomicAdd(&accum[T_LEN + t], den);
}

__global__ void loss_kernel(const float* __restrict__ accum, float* __restrict__ out)
{
    if (threadIdx.x == 0) {
        float s = 0.0f;
        for (int t = 0; t < T_LEN; ++t) s += accum[t] / (accum[T_LEN + t] + 1e-5f);
        out[0] = s * 0.3f;
    }
}

// ---------------------------------------------------------------------------
extern "C" void kernel_launch(void* const* d_in, const int* in_sizes, int n_in,
                              void* d_out, int out_size, void* d_ws, size_t ws_size,
                              hipStream_t stream)
{
    const float* fwd_values     = (const float*)d_in[0];
    const float* fwd_masks      = (const float*)d_in[1];
    const float* fwd_deltas     = (const float*)d_in[2];
    const float* fwd_evals      = (const float*)d_in[3];
    const float* fwd_eval_masks = (const float*)d_in[4];
    const float* bwd_values     = (const float*)d_in[5];
    const float* bwd_masks      = (const float*)d_in[6];
    const float* bwd_deltas     = (const float*)d_in[7];
    const float* W_ih   = (const float*)d_in[8];
    const float* W_hh   = (const float*)d_in[9];
    const float* b_ih   = (const float*)d_in[10];
    const float* b_hh   = (const float*)d_in[11];
    const float* W_hist = (const float*)d_in[12];
    const float* b_hist = (const float*)d_in[13];
    const float* W_wc   = (const float*)d_in[14];
    const float* b_wc   = (const float*)d_in[15];

    char* ws = (char*)d_ws;
    float*  xhf   = (float*)(ws);
    float*  xhb   = (float*)(ws + (size_t)BTF_ * 4);
    __bf16* Wcat  = (__bf16*)(ws + (size_t)BTF_ * 8);
    float*  biasc = (float*)(ws + (size_t)BTF_ * 8 + (size_t)N4H * KPAD * 2);
    __bf16* Whf   = (__bf16*)((char*)biasc + N4H * 4);
    __bf16* Whb   = Whf + 48 * HID;
    float*  accum = (float*)(Whb + 48 * HID);

    prep_kernel<<<512, 256, 0, stream>>>(W_ih, W_hh, b_ih, b_hh, W_hist,
                                         Wcat, biasc, Whf, Whb, accum);
    lstm_kernel<<<BATCH / BT, NTHR, 0, stream>>>(fwd_values, fwd_masks, fwd_deltas,
                                                 Wcat, biasc, Whf, xhf, 0);
    lstm_kernel<<<BATCH / BT, NTHR, 0, stream>>>(bwd_values, bwd_masks, bwd_deltas,
                                                 Wcat, biasc, Whb, xhb, 1);
    finalize_kernel<<<(BATCH * T_LEN + 255) / 256, 256, 0, stream>>>(
        fwd_values, fwd_masks, fwd_deltas, fwd_evals, fwd_eval_masks,
        xhf, xhb, W_wc, b_wc, b_hist, (float*)d_out, accum);
    loss_kernel<<<1, 32, 0, stream>>>(accum, (float*)d_out);
}
